// GCNModel_77704548319367
// MI455X (gfx1250) — compile-verified, matrix operand order checked by reading, not run on hardware
//
#include <hip/hip_runtime.h>
#include <hip/hip_bf16.h>
#include <stdint.h>

// ---------------------------------------------------------------------------
// Types / helpers
// ---------------------------------------------------------------------------
typedef __attribute__((ext_vector_type(16))) __bf16 v16bf;
typedef __attribute__((ext_vector_type(8)))  __bf16 v8bf;
typedef __attribute__((ext_vector_type(8)))  float  v8f;

__device__ __forceinline__ unsigned short f2bf(float f) {
    // round-to-nearest-even fp32 -> bf16
    unsigned int u = __float_as_uint(f);
    u = u + 0x7FFFu + ((u >> 16) & 1u);
    return (unsigned short)(u >> 16);
}
// monotonic float<->int encoding for atomicMax on floats
__device__ __forceinline__ int f2ord(float f) {
    int i = __float_as_int(f);
    return i < 0 ? (i ^ 0x7FFFFFFF) : i;
}
__device__ __forceinline__ float ord2f(int i) {
    return __int_as_float(i < 0 ? (i ^ 0x7FFFFFFF) : i);
}

// ---------------------------------------------------------------------------
// WMMA bf16 GEMM: C[M,N] = A[M,K] @ B[K,N] (+bias1 +bias2), f32 accumulate.
// Requires: K % 32 == 0, N % BN == 0, BN in {64,128,256}. M guarded.
// Block tile 64 x BN; BN threads (2*BN/64 waves); each wave owns 32x64
// (2x4 WMMA fragments -> 8 v_wmma per K-step).
// ---------------------------------------------------------------------------
template<int BN>
__global__ __launch_bounds__(256) void k_gemm_bf16(
    const float* __restrict__ A, const float* __restrict__ B,
    float* __restrict__ C, int M, int N, int K,
    const float* __restrict__ bias1, const float* __restrict__ bias2)
{
    constexpr int NT = BN;         // threads per block
    constexpr int WN = BN / 64;    // waves along N
    __shared__ unsigned short As[64 * 32];   // [m][k]
    __shared__ unsigned short Bs[BN * 32];   // [n][k] (B transposed)

    const int tid  = threadIdx.x;
    const int lane = tid & 31;
    const int wave = tid >> 5;
    const int wm   = (wave / WN) * 32;   // wave row offset (0 or 32)
    const int wn0  = (wave % WN) * 64;   // wave col offset
    const int bm0  = blockIdx.y * 64;
    const int bn0  = blockIdx.x * BN;

    const v8f vzero = {0.f, 0.f, 0.f, 0.f, 0.f, 0.f, 0.f, 0.f};
    v8f acc[2][4];
    #pragma unroll
    for (int i = 0; i < 2; ++i)
        #pragma unroll
        for (int j = 0; j < 4; ++j) acc[i][j] = vzero;

    const int r  = lane & 15;   // row (A) / col (B) within fragment
    const int kh = lane >> 4;   // K-half selector (ISA 16-bit operand layout)

    for (int kt = 0; kt < K; kt += 32) {
        // --- cooperative load A tile 64x32 (fp32 -> bf16) ---
        #pragma unroll
        for (int j = 0; j < 512 / NT; ++j) {
            int linear = tid + j * NT;            // 512 float4's
            int row = linear >> 3;                // 8 float4 per 32-wide row
            int col = (linear & 7) << 2;
            float4 v = make_float4(0.f, 0.f, 0.f, 0.f);
            int gr = bm0 + row;
            if (gr < M) v = *(const float4*)(A + (size_t)gr * K + kt + col);
            unsigned short* p = &As[row * 32 + col];
            p[0] = f2bf(v.x); p[1] = f2bf(v.y); p[2] = f2bf(v.z); p[3] = f2bf(v.w);
        }
        // --- cooperative load B tile 32xBN, store transposed [n][k] ---
        #pragma unroll
        for (int j = 0; j < 8; ++j) {
            int linear = tid + j * NT;            // 8*BN float4's
            int kk = linear / (BN / 4);
            int nn = (linear % (BN / 4)) << 2;
            float4 v = *(const float4*)(B + (size_t)(kt + kk) * N + bn0 + nn);
            Bs[(nn + 0) * 32 + kk] = f2bf(v.x);
            Bs[(nn + 1) * 32 + kk] = f2bf(v.y);
            Bs[(nn + 2) * 32 + kk] = f2bf(v.z);
            Bs[(nn + 3) * 32 + kk] = f2bf(v.w);
        }
        __syncthreads();

        // --- B fragments for this wave's 4 column tiles ---
        v16bf bfr[4];
        #pragma unroll
        for (int j = 0; j < 4; ++j) {
            const unsigned short* bp = &Bs[(wn0 + j * 16 + r) * 32 + kh * 8];
            v8bf blo = *(const v8bf*)(bp);
            v8bf bhi = *(const v8bf*)(bp + 16);
            bfr[j] = __builtin_shufflevector(blo, bhi,
                      0, 1, 2, 3, 4, 5, 6, 7, 8, 9, 10, 11, 12, 13, 14, 15);
        }
        // --- 2x4 WMMA per wave ---
        #pragma unroll
        for (int i = 0; i < 2; ++i) {
            const unsigned short* ap = &As[(wm + i * 16 + r) * 32 + kh * 8];
            v8bf alo = *(const v8bf*)(ap);
            v8bf ahi = *(const v8bf*)(ap + 16);
            v16bf av = __builtin_shufflevector(alo, ahi,
                        0, 1, 2, 3, 4, 5, 6, 7, 8, 9, 10, 11, 12, 13, 14, 15);
            #pragma unroll
            for (int j = 0; j < 4; ++j) {
                acc[i][j] = __builtin_amdgcn_wmma_f32_16x16x32_bf16(
                    false, av, false, bfr[j], (short)0, acc[i][j], false, false);
            }
        }
        __syncthreads();
    }

    // --- epilogue: C/D layout: lane = N col (0-15), vgpr t -> M row t + 8*(lane>>4)
    #pragma unroll
    for (int i = 0; i < 2; ++i) {
        #pragma unroll
        for (int j = 0; j < 4; ++j) {
            int col = bn0 + wn0 + j * 16 + (lane & 15);
            float bv = 0.f;
            if (bias1) bv += bias1[col];
            if (bias2) bv += bias2[col];
            int rbase = bm0 + wm + i * 16 + (lane >> 4) * 8;
            #pragma unroll
            for (int t = 0; t < 8; ++t) {
                int rr = rbase + t;
                if (rr < M) C[(size_t)rr * N + col] = acc[i][j][t] + bv;
            }
        }
    }
}

// ---------------------------------------------------------------------------
// Fills
// ---------------------------------------------------------------------------
__global__ void k_fill_f32(float* p, float v, long long n) {
    long long i = blockIdx.x * (long long)blockDim.x + threadIdx.x;
    if (i < n) p[i] = v;
}
__global__ void k_fill_i32(int* p, int v, long long n) {
    long long i = blockIdx.x * (long long)blockDim.x + threadIdx.x;
    if (i < n) p[i] = v;
}

// ---------------------------------------------------------------------------
// Degree / inv-sqrt-degree (shared by both GCN layers; self-loops included)
// ---------------------------------------------------------------------------
__global__ void k_deg(const int* __restrict__ ei, int E, int N, float* __restrict__ deg) {
    int e = blockIdx.x * blockDim.x + threadIdx.x;
    if (e >= E + N) return;
    int d = (e < E) ? ei[E + e] : (e - E);
    atomicAdd(&deg[d], 1.0f);
}
__global__ void k_dinv(const float* __restrict__ deg, float* __restrict__ dinv, int N) {
    int n = blockIdx.x * blockDim.x + threadIdx.x;
    if (n >= N) return;
    float d = deg[n];
    dinv[n] = d > 0.f ? rsqrtf(d) : 0.f;
}

// ---------------------------------------------------------------------------
// GAT attention logits: es[n,h] = <h[n,h,:], a_src[h,:]>, ed analogous
// ---------------------------------------------------------------------------
__global__ void k_att(const float* __restrict__ hfeat,
                      const float* __restrict__ a_src, const float* __restrict__ a_dst,
                      float* __restrict__ es, float* __restrict__ ed, int N, int Cc)
{
    int i = blockIdx.x * blockDim.x + threadIdx.x;
    if (i >= N * 8) return;
    int n = i >> 3, h = i & 7;
    const float* hp = hfeat + (size_t)n * 8 * Cc + h * Cc;
    const float* av = a_src + h * Cc;
    const float* bv = a_dst + h * Cc;
    float s = 0.f, d = 0.f;
    for (int c = 0; c < Cc; ++c) { float v = hp[c]; s += v * av[c]; d += v * bv[c]; }
    es[i] = s; ed[i] = d;
}

// ---------------------------------------------------------------------------
// Segment-softmax passes (H = 8, leaky_relu slope 0.2)
// ---------------------------------------------------------------------------
__global__ void k_edge_max(const int* __restrict__ ei, int E, int N,
                           const float* __restrict__ es, const float* __restrict__ ed,
                           int* __restrict__ emax)
{
    int e = blockIdx.x * blockDim.x + threadIdx.x;
    if (e >= E + N) return;
    int s, d;
    if (e < E) { s = ei[e]; d = ei[E + e]; } else { s = d = e - E; }
    #pragma unroll
    for (int h = 0; h < 8; ++h) {
        float v = es[s * 8 + h] + ed[d * 8 + h];
        v = v < 0.f ? 0.2f * v : v;
        atomicMax(&emax[d * 8 + h], f2ord(v));
    }
}
__global__ void k_edge_exp(const int* __restrict__ ei, int E, int N,
                           const float* __restrict__ es, const float* __restrict__ ed,
                           const int* __restrict__ emax,
                           float* __restrict__ alpha, float* __restrict__ den)
{
    int e = blockIdx.x * blockDim.x + threadIdx.x;
    if (e >= E + N) return;
    int s, d;
    if (e < E) { s = ei[e]; d = ei[E + e]; } else { s = d = e - E; }
    #pragma unroll
    for (int h = 0; h < 8; ++h) {
        float v = es[s * 8 + h] + ed[d * 8 + h];
        v = v < 0.f ? 0.2f * v : v;
        float ex = __expf(v - ord2f(emax[d * 8 + h]));
        alpha[(size_t)e * 8 + h] = ex;
        atomicAdd(&den[d * 8 + h], ex);
    }
}
// normalize: alpha[e,h] /= (den[dst,h] + 1e-16)  (once per edge/head)
__global__ void k_alpha_norm(const int* __restrict__ ei, int E, int N,
                             const float* __restrict__ den, float* __restrict__ alpha)
{
    int e = blockIdx.x * blockDim.x + threadIdx.x;
    if (e >= E + N) return;
    int d = (e < E) ? ei[E + e] : (e - E);
    #pragma unroll
    for (int h = 0; h < 8; ++h)
        alpha[(size_t)e * 8 + h] /= (den[d * 8 + h] + 1e-16f);
}
// weighted scatter: acc[dst, h, c] += alpha[e,h] * hfeat[src, h, c]
__global__ void k_scatter_gat(const int* __restrict__ ei, int E, int N,
                              const float* __restrict__ hfeat,
                              const float* __restrict__ alpha,
                              float* __restrict__ acc, int Cc)
{
    int HC = 8 * Cc;
    int groups = HC >> 2;
    long long idx = blockIdx.x * (long long)blockDim.x + threadIdx.x;
    long long total = (long long)(E + N) * groups;
    if (idx >= total) return;
    int e = (int)(idx / groups);
    int g = (int)(idx - (long long)e * groups);
    int c0 = g << 2;
    int h = c0 / Cc;
    int s, d;
    if (e < E) { s = ei[e]; d = ei[E + e]; } else { s = d = e - E; }
    float a = alpha[(size_t)e * 8 + h];
    float4 v = *(const float4*)(hfeat + (size_t)s * HC + c0);
    float* p = acc + (size_t)d * HC + c0;
    atomicAdd(p + 0, a * v.x); atomicAdd(p + 1, a * v.y);
    atomicAdd(p + 2, a * v.z); atomicAdd(p + 3, a * v.w);
}

// ---------------------------------------------------------------------------
// GCN scatter: acc[dst, c] += dinv[src]*dinv[dst] * hfeat[src, c]
// ---------------------------------------------------------------------------
__global__ void k_scatter_gcn(const int* __restrict__ ei, int E, int N,
                              const float* __restrict__ hfeat, const float* __restrict__ dinv,
                              float* __restrict__ acc, int C)
{
    int groups = C >> 2;
    long long idx = blockIdx.x * (long long)blockDim.x + threadIdx.x;
    long long total = (long long)(E + N) * groups;
    if (idx >= total) return;
    int e = (int)(idx / groups);
    int c0 = (int)(idx - (long long)e * groups) << 2;
    int s, d;
    if (e < E) { s = ei[e]; d = ei[E + e]; } else { s = d = e - E; }
    float norm = dinv[s] * dinv[d];
    float4 v = *(const float4*)(hfeat + (size_t)s * C + c0);
    float* p = acc + (size_t)d * C + c0;
    atomicAdd(p + 0, norm * v.x); atomicAdd(p + 1, norm * v.y);
    atomicAdd(p + 2, norm * v.z); atomicAdd(p + 3, norm * v.w);
}

// ---------------------------------------------------------------------------
// LayerNorm + ReLU: one wave32 per node
// ---------------------------------------------------------------------------
__global__ __launch_bounds__(256) void k_ln_relu(
    const float* __restrict__ in, float* __restrict__ out,
    const float* __restrict__ g, const float* __restrict__ b, int N, int C)
{
    int wave = threadIdx.x >> 5, lane = threadIdx.x & 31;
    int n = blockIdx.x * 8 + wave;
    if (n >= N) return;
    const float* row = in + (size_t)n * C;
    float s = 0.f, s2 = 0.f;
    for (int c = lane; c < C; c += 32) { float v = row[c]; s += v; s2 += v * v; }
    #pragma unroll
    for (int off = 16; off; off >>= 1) {
        s  += __shfl_xor(s, off, 32);
        s2 += __shfl_xor(s2, off, 32);
    }
    float m   = s / C;
    float var = s2 / C - m * m;
    float rs  = rsqrtf(var + 1e-5f);
    for (int c = lane; c < C; c += 32) {
        float v = (row[c] - m) * rs * g[c] + b[c];
        out[(size_t)n * C + c] = v > 0.f ? v : 0.f;
    }
}

// ---------------------------------------------------------------------------
// GAT3 head-mean: out[n,c] = mean_h acc[n,h,c] + mlp[n,c] + bias[c]
// ---------------------------------------------------------------------------
__global__ void k_meanheads(const float* __restrict__ acc, const float* __restrict__ mlp,
                            const float* __restrict__ bias, float* __restrict__ out, int N)
{
    int i = blockIdx.x * blockDim.x + threadIdx.x;
    if (i >= N * 64) return;
    int n = i >> 6, c = i & 63;
    float s = 0.f;
    #pragma unroll
    for (int h = 0; h < 8; ++h) s += acc[(size_t)n * 512 + h * 64 + c];
    out[i] = 0.125f * s + mlp[i] + bias[c];
}

// ---------------------------------------------------------------------------
// Layer 4 (N_out=2): per-node dual projection; out init = h@Wm4+bm4+b4g
// ---------------------------------------------------------------------------
__global__ void k_final_linear(const float* __restrict__ h,
                               const float* __restrict__ W4g, const float* __restrict__ b4g,
                               const float* __restrict__ Wm4, const float* __restrict__ bm4,
                               float* __restrict__ out, float* __restrict__ hfeat4, int N)
{
    int n = blockIdx.x * blockDim.x + threadIdx.x;
    if (n >= N) return;
    float o0 = b4g[0] + bm4[0], o1 = b4g[1] + bm4[1];
    float f0 = 0.f, f1 = 0.f;
    const float* hp = h + (size_t)n * 64;
    #pragma unroll
    for (int c = 0; c < 64; ++c) {
        float v = hp[c];
        f0 += v * W4g[c * 2 + 0]; f1 += v * W4g[c * 2 + 1];
        o0 += v * Wm4[c * 2 + 0]; o1 += v * Wm4[c * 2 + 1];
    }
    out[n * 2 + 0] = o0; out[n * 2 + 1] = o1;
    hfeat4[n * 2 + 0] = f0; hfeat4[n * 2 + 1] = f1;
}
__global__ void k_scatter4(const int* __restrict__ ei, int E, int N,
                           const float* __restrict__ hfeat4, const float* __restrict__ dinv,
                           float* __restrict__ out)
{
    int e = blockIdx.x * blockDim.x + threadIdx.x;
    if (e >= E + N) return;
    int s, d;
    if (e < E) { s = ei[e]; d = ei[E + e]; } else { s = d = e - E; }
    float norm = dinv[s] * dinv[d];
    atomicAdd(&out[d * 2 + 0], norm * hfeat4[s * 2 + 0]);
    atomicAdd(&out[d * 2 + 1], norm * hfeat4[s * 2 + 1]);
}

// ---------------------------------------------------------------------------
// Host launch
// ---------------------------------------------------------------------------
extern "C" void kernel_launch(void* const* d_in, const int* in_sizes, int n_in,
                              void* d_out, int out_size, void* d_ws, size_t ws_size,
                              hipStream_t stream)
{
    (void)n_in; (void)out_size; (void)ws_size;
    const float* x   = (const float*)d_in[0];
    const int*   ei  = (const int*)d_in[1];
    const float* W1g = (const float*)d_in[2];
    const float* as1 = (const float*)d_in[3];
    const float* ad1 = (const float*)d_in[4];
    const float* b1g = (const float*)d_in[5];
    const float* W2g = (const float*)d_in[6];
    const float* b2g = (const float*)d_in[7];
    const float* W3g = (const float*)d_in[8];
    const float* as3 = (const float*)d_in[9];
    const float* ad3 = (const float*)d_in[10];
    const float* b3g = (const float*)d_in[11];
    const float* W4g = (const float*)d_in[12];
    const float* b4g = (const float*)d_in[13];
    const float* Wm1 = (const float*)d_in[14];
    const float* bm1 = (const float*)d_in[15];
    const float* Wm2 = (const float*)d_in[16];
    const float* bm2 = (const float*)d_in[17];
    const float* Wm3 = (const float*)d_in[18];
    const float* bm3 = (const float*)d_in[19];
    const float* Wm4 = (const float*)d_in[20];
    const float* bm4 = (const float*)d_in[21];
    const float* g0  = (const float*)d_in[22];
    const float* be0 = (const float*)d_in[23];
    const float* g1w = (const float*)d_in[24];
    const float* be1 = (const float*)d_in[25];
    const float* g2w = (const float*)d_in[26];
    const float* be2 = (const float*)d_in[27];

    const int N  = in_sizes[0] / 1024;
    const int E  = in_sizes[1] / 2;
    const int EP = E + N;

    // workspace layout (floats)
    float* ws = (float*)d_ws;
    size_t off = 0;
    float* hfeat = ws + off; off += (size_t)N * 512;   // conv features (max N x 512)
    float* accb  = ws + off; off += (size_t)N * 512;   // aggregation accumulator
    float* h1    = ws + off; off += (size_t)N * 256;
    float* h2    = ws + off; off += (size_t)N * 128;
    float* h3    = ws + off; off += (size_t)N * 64;
    float* mlp3  = ws + off; off += (size_t)N * 64;
    float* esb   = ws + off; off += (size_t)N * 8;
    float* edb   = ws + off; off += (size_t)N * 8;
    int*   emax  = (int*)(ws + off); off += (size_t)N * 8;
    float* den   = ws + off; off += (size_t)N * 8;
    float* alpha = ws + off; off += (size_t)EP * 8;
    float* deg   = ws + off; off += (size_t)N;
    float* dinv  = ws + off; off += (size_t)N;

    auto cdiv = [](long long a, long long b) { return (unsigned int)((a + b - 1) / b); };
    const int T = 256;
    const int ORD_NEG_INF = 0x807FFFFF;   // f2ord(-inf)

    // degree (identical graph for both GCN layers)
    k_fill_f32<<<cdiv(N, T), T, 0, stream>>>(deg, 0.f, N);
    k_deg<<<cdiv(EP, T), T, 0, stream>>>(ei, E, N, deg);
    k_dinv<<<cdiv(N, T), T, 0, stream>>>(deg, dinv, N);

    // ---- Layer 1: GAT(1024 -> 8x32 concat) + x@Wm1, LN, ReLU ----
    dim3 gL1(1, cdiv(N, 64));
    k_gemm_bf16<256><<<gL1, 256, 0, stream>>>(x, W1g, hfeat, N, 256, 1024, nullptr, nullptr);
    k_gemm_bf16<256><<<gL1, 256, 0, stream>>>(x, Wm1, accb, N, 256, 1024, bm1, b1g);
    k_att<<<cdiv((long long)N * 8, T), T, 0, stream>>>(hfeat, as1, ad1, esb, edb, N, 32);
    k_fill_i32<<<cdiv((long long)N * 8, T), T, 0, stream>>>(emax, ORD_NEG_INF, (long long)N * 8);
    k_fill_f32<<<cdiv((long long)N * 8, T), T, 0, stream>>>(den, 0.f, (long long)N * 8);
    k_edge_max<<<cdiv(EP, T), T, 0, stream>>>(ei, E, N, esb, edb, emax);
    k_edge_exp<<<cdiv(EP, T), T, 0, stream>>>(ei, E, N, esb, edb, emax, alpha, den);
    k_alpha_norm<<<cdiv(EP, T), T, 0, stream>>>(ei, E, N, den, alpha);
    k_scatter_gat<<<cdiv((long long)EP * 64, T), T, 0, stream>>>(ei, E, N, hfeat, alpha, accb, 32);
    k_ln_relu<<<cdiv(N, 8), 256, 0, stream>>>(accb, h1, g0, be0, N, 256);

    // ---- Layer 2: GCN(256->128) + h1@Wm2, LN, ReLU ----
    dim3 gL2(1, cdiv(N, 64));
    k_gemm_bf16<128><<<gL2, 128, 0, stream>>>(h1, W2g, hfeat, N, 128, 256, nullptr, nullptr);
    k_gemm_bf16<128><<<gL2, 128, 0, stream>>>(h1, Wm2, accb, N, 128, 256, bm2, b2g);
    k_scatter_gcn<<<cdiv((long long)EP * 32, T), T, 0, stream>>>(ei, E, N, hfeat, dinv, accb, 128);
    k_ln_relu<<<cdiv(N, 8), 256, 0, stream>>>(accb, h2, g1w, be1, N, 128);

    // ---- Layer 3: GAT(128 -> 8x64, head-mean) + h2@Wm3, LN, ReLU ----
    dim3 gL3(2, cdiv(N, 64));
    k_gemm_bf16<256><<<gL3, 256, 0, stream>>>(h2, W3g, hfeat, N, 512, 128, nullptr, nullptr);
    dim3 gL3m(1, cdiv(N, 64));
    k_gemm_bf16<64><<<gL3m, 64, 0, stream>>>(h2, Wm3, mlp3, N, 64, 128, bm3, nullptr);
    k_att<<<cdiv((long long)N * 8, T), T, 0, stream>>>(hfeat, as3, ad3, esb, edb, N, 64);
    k_fill_i32<<<cdiv((long long)N * 8, T), T, 0, stream>>>(emax, ORD_NEG_INF, (long long)N * 8);
    k_fill_f32<<<cdiv((long long)N * 8, T), T, 0, stream>>>(den, 0.f, (long long)N * 8);
    k_edge_max<<<cdiv(EP, T), T, 0, stream>>>(ei, E, N, esb, edb, emax);
    k_edge_exp<<<cdiv(EP, T), T, 0, stream>>>(ei, E, N, esb, edb, emax, alpha, den);
    k_alpha_norm<<<cdiv(EP, T), T, 0, stream>>>(ei, E, N, den, alpha);
    k_fill_f32<<<cdiv((long long)N * 512, T), T, 0, stream>>>(accb, 0.f, (long long)N * 512);
    k_scatter_gat<<<cdiv((long long)EP * 128, T), T, 0, stream>>>(ei, E, N, hfeat, alpha, accb, 64);
    k_meanheads<<<cdiv((long long)N * 64, T), T, 0, stream>>>(accb, mlp3, b3g, hfeat, N);
    k_ln_relu<<<cdiv(N, 8), 256, 0, stream>>>(hfeat, h3, g2w, be2, N, 64);

    // ---- Layer 4: GCN(64->2) + h3@Wm4 -> d_out ----
    float* out = (float*)d_out;
    k_final_linear<<<cdiv(N, T), T, 0, stream>>>(h3, W4g, b4g, Wm4, bm4, out, accb, N);
    k_scatter4<<<cdiv(EP, T), T, 0, stream>>>(ei, E, N, accb, dinv, out);
}
